// AttentionMap_32796370272786
// MI455X (gfx1250) — matrix operand order repaired and checked by reading, not compiled, not run on hardware
//
#include <hip/hip_runtime.h>
#include <math.h>

// ---------------- problem constants (from reference) ----------------
#define Bv   8
#define CIN  64
#define COUT 8
#define HIN  128
#define WIN  128
#define H2   256      // upsampled H/W
#define W2   256
#define UPH  258      // upsampled + 1px zero halo for 3x3 conv
#define UPW  258
#define XPH  260      // xf padded by 2 for window ops / patch extraction
#define XPW  260
#define NUMK 8        // top-k
#define KWIN 5

#define PXPITCH 72    // LDS pixel pitch in ushorts (64 data + 8 pad) -> 144B, bank-conflict free

// ---------------- WMMA fragment types (gfx1250, wave32) ----------------
typedef __attribute__((ext_vector_type(16))) __bf16        v16bf;
typedef __attribute__((ext_vector_type(8)))  float         v8f;
typedef __attribute__((ext_vector_type(8)))  unsigned short u16x8;

union FragBF {            // 32 bytes: two b128 chunks viewed as one WMMA operand
    u16x8 h[2];
    v16bf v;
};

__device__ __forceinline__ unsigned short f2bf(float f) {
    unsigned int u = __builtin_bit_cast(unsigned int, f);
    unsigned int r = u + 0x7FFFu + ((u >> 16) & 1u);   // round-to-nearest-even
    return (unsigned short)(r >> 16);
}

// LDS byte offset of a __shared__ object (ptrtoint of an AS(3) pointer)
__device__ __forceinline__ unsigned lds_off_of(const unsigned short* p) {
    return (unsigned)(unsigned long long)
        (__attribute__((address_space(3))) const unsigned short*)p;
}

// ---------------- prep: fold BN, swizzle weights into WMMA-B layout ----------------
// wp1 layout: [tap(9)][ctile(4)][kchunk(2)][lane(32)][el(16)]  (bf16 raw)
//   lane L holds column co = ctile*16 + (L&15); K = kchunk*32 + (el<8 ? (L>>4)*8 + el : 16 + (L>>4)*8 + el-8)
// wp2 layout: [kchunk(2)][lane(32)][el(16)], co = L&15 (>=8 zero-padded)
__global__ __launch_bounds__(256) void prep_kernel(
    const float* __restrict__ w1, const float* __restrict__ b1,
    const float* __restrict__ g1, const float* __restrict__ be1,
    const float* __restrict__ m1, const float* __restrict__ v1,
    const float* __restrict__ w2, const float* __restrict__ g2,
    const float* __restrict__ be2, const float* __restrict__ m2,
    const float* __restrict__ v2,
    unsigned short* __restrict__ wp1, unsigned short* __restrict__ wp2,
    float* __restrict__ s1, float* __restrict__ sh1,
    float* __restrict__ s2, float* __restrict__ sh2)
{
    int i = blockIdx.x * 256 + threadIdx.x;
    const int N1 = 9 * 4 * 2 * 32 * 16;     // 36864
    const int N2 = 2 * 32 * 16;             // 1024
    if (i < N1) {
        int t = i / 4096; int r = i % 4096;
        int ct = r / 1024; r %= 1024;
        int kc = r / 512;  r %= 512;
        int lane = r / 16; int el = r % 16;
        int co = ct * 16 + (lane & 15);
        int ci = kc * 32 + (el < 8 ? (lane >> 4) * 8 + el : 16 + (lane >> 4) * 8 + (el - 8));
        int ky = t / 3, kx = t % 3;
        wp1[i] = f2bf(w1[((co * CIN + ci) * 3 + ky) * 3 + kx]);   // OIHW
    } else if (i < N1 + N2) {
        int j = i - N1;
        int kc = j / 512; int r = j % 512;
        int lane = r / 16; int el = r % 16;
        int co = lane & 15;
        int ci = kc * 32 + (el < 8 ? (lane >> 4) * 8 + el : 16 + (lane >> 4) * 8 + (el - 8));
        wp2[j] = (co < COUT) ? f2bf(w2[co * CIN + ci]) : (unsigned short)0;
    } else if (i < N1 + N2 + CIN) {
        int c = i - N1 - N2;
        float sc = g1[c] * rsqrtf(v1[c] + 1e-5f);
        s1[c]  = sc;
        sh1[c] = be1[c] + sc * (b1[c] - m1[c]);    // BN(conv + bias)
    } else if (i < N1 + N2 + CIN + COUT) {
        int c = i - N1 - N2 - CIN;
        float sc = g2[c] * rsqrtf(v2[c] + 1e-5f);
        s2[c]  = sc;
        sh2[c] = be2[c] - sc * m2[c];              // 1x1 has no bias
    }
}

// ---------------- bilinear 2x upsample (half-pixel) -> NHWC bf16 with 1px zero halo ----
__global__ __launch_bounds__(256) void upsample_kernel(
    const float* __restrict__ x, unsigned short* __restrict__ up)
{
    long i = (long)blockIdx.x * 256 + threadIdx.x;
    const long n = (long)Bv * UPH * UPW * CIN;
    if (i >= n) return;
    int c  = (int)(i & 63);
    long r = i >> 6;
    int xp = (int)(r % UPW); r /= UPW;
    int yp = (int)(r % UPH);
    int b  = (int)(r / UPH);
    unsigned short o = 0;
    if (xp > 0 && xp < UPW - 1 && yp > 0 && yp < UPH - 1) {
        int xo = xp - 1, yo = yp - 1;
        float sx = 0.5f * xo - 0.25f;  sx = sx < 0.f ? 0.f : sx;
        float sy = 0.5f * yo - 0.25f;  sy = sy < 0.f ? 0.f : sy;
        int ix0 = (int)sx, iy0 = (int)sy;
        int ix1 = ix0 + 1 < WIN ? ix0 + 1 : WIN - 1;
        int iy1 = iy0 + 1 < HIN ? iy0 + 1 : HIN - 1;
        float fx = sx - ix0, fy = sy - iy0;
        const float* p = x + (long)(b * CIN + c) * HIN * WIN;
        float v00 = p[iy0 * WIN + ix0], v01 = p[iy0 * WIN + ix1];
        float v10 = p[iy1 * WIN + ix0], v11 = p[iy1 * WIN + ix1];
        float v = (1.f - fy) * ((1.f - fx) * v00 + fx * v01)
                +        fy  * ((1.f - fx) * v10 + fx * v11);
        o = f2bf(v);
    }
    up[i] = o;
}

// ---------------- fused conv3x3(WMMA bf16) + BN1 + ReLU + conv1x1(WMMA) + BN2 ----------
// Workgroup = 8 waves: wave = (pt, ct) -> 2 pixel-tiles x 4 cout-tiles (32px x 64ch).
// Activations for the 3 needed rows are DMA'd into LDS once (async global->LDS),
// then A fragments come from LDS (ds_load_b128, conflict-free 144B pixel pitch).
__global__ __launch_bounds__(256) void conv_fused_kernel(
    const unsigned short* __restrict__ up,
    const unsigned short* __restrict__ wp1,
    const unsigned short* __restrict__ wp2,
    const float* __restrict__ s1, const float* __restrict__ sh1,
    const float* __restrict__ s2, const float* __restrict__ sh2,
    float* __restrict__ yout)
{
    __shared__ __align__(16) unsigned short stage[3 * 34 * PXPITCH]; // 3 rows x 34 px
    __shared__ __align__(16) unsigned short lsd[2 * 16 * PXPITCH];   // [pt][m][ci] bf16

    int wg = blockIdx.x;
    int tx = wg & 7;               // 8 tiles of 32px per row
    int y  = (wg >> 3) & 255;
    int b  = wg >> 11;
    int lane  = threadIdx.x & 31;
    int wave  = threadIdx.x >> 5;
    int pt    = wave >> 2;
    int ct    = wave & 3;
    int khalf = lane >> 4;         // lane half selects K sub-stripes (ISA A layout)
    int ln    = lane & 15;
    int xg = tx * 32;              // workgroup x origin
    int x0 = xg + pt * 16;

    // warm L2 for the row the *next* workgroup will stage
    {
        int yn = y + 3 <= UPH - 1 ? y + 3 : UPH - 1;
        __builtin_prefetch(up + ((long)(b * UPH + yn) * UPW + xg + lane) * 64, 0, 0);
    }

    // ---- async DMA: 3 rows x 34 px x 128B -> LDS (816 x b128 transfers) ----
    {
        unsigned base_lds = lds_off_of(stage);
        const int total = 3 * 34 * 8;
        for (int k = threadIdx.x; k < total; k += 256) {
            int row = k / 272;
            int rem = k - row * 272;
            int px  = rem >> 3;
            int ch  = rem & 7;
            unsigned goff = (unsigned)((((b * UPH + y + row) * UPW + (xg + px)) << 6)
                                       + (ch << 3)) * 2u;                     // bytes
            unsigned loff = base_lds
                          + (unsigned)((row * 34 + px) * PXPITCH + (ch << 3)) * 2u;
            asm volatile("global_load_async_to_lds_b128 %0, %1, %2"
                         :: "v"(loff), "v"(goff), "s"(up) : "memory");
        }
        asm volatile("s_wait_asynccnt 0" ::: "memory");
        __syncthreads();
    }

    v8f acc = {0.f, 0.f, 0.f, 0.f, 0.f, 0.f, 0.f, 0.f};
    #pragma unroll
    for (int t = 0; t < 9; ++t) {
        int ky = t / 3, kx = t % 3;
        // A: 16 pixels x 32 cin (two chunks) from LDS stage
        const unsigned short* ap =
            stage + (ky * 34 + (pt * 16 + ln + kx)) * PXPITCH + (khalf << 3);
        FragBF a0, a1;
        a0.h[0] = *(const u16x8*)(ap);        // cin  0.. 7 (+8 on hi half)
        a0.h[1] = *(const u16x8*)(ap + 16);   // cin 16..23 (+8)
        a1.h[0] = *(const u16x8*)(ap + 32);   // cin 32..39 (+8)
        a1.h[1] = *(const u16x8*)(ap + 48);   // cin 48..55 (+8)
        // B: pre-swizzled weights, contiguous per lane (L2-resident)
        const unsigned short* bp = wp1 + ((t * 4 + ct) * 2) * 512 + lane * 16;
        FragBF b0, b1;
        b0.h[0] = *(const u16x8*)(bp);
        b0.h[1] = *(const u16x8*)(bp + 8);
        b1.h[0] = *(const u16x8*)(bp + 512);
        b1.h[1] = *(const u16x8*)(bp + 520);
        acc = __builtin_amdgcn_wmma_f32_16x16x32_bf16(false, a0.v, false, b0.v,
                                                      (short)0, acc, false, false);
        acc = __builtin_amdgcn_wmma_f32_16x16x32_bf16(false, a1.v, false, b1.v,
                                                      (short)0, acc, false, false);
    }

    // BN1 + ReLU -> LDS bf16 in A-layout-friendly [pt][m][ci]
    int co = ct * 16 + ln;
    float sc = s1[co], sf = sh1[co];
    #pragma unroll
    for (int j = 0; j < 8; ++j) {
        int m = j + (khalf << 3);                      // C-layout row
        float v = fmaf(acc[j], sc, sf);
        v = v > 0.f ? v : 0.f;
        lsd[((pt << 4) + m) * PXPITCH + co] = f2bf(v);
    }
    __syncthreads();

    // 1x1 conv: one wave per pixel-tile (ct==0). M=16 px, N=16 (co<8 valid), K=64.
    if (ct == 0) {
        const unsigned short* lp = lsd + ((pt << 4) + ln) * PXPITCH + (khalf << 3);
        FragBF a0, a1;
        a0.h[0] = *(const u16x8*)(lp);
        a0.h[1] = *(const u16x8*)(lp + 16);
        a1.h[0] = *(const u16x8*)(lp + 32);
        a1.h[1] = *(const u16x8*)(lp + 48);
        const unsigned short* bp = wp2 + lane * 16;
        FragBF b0, b1;
        b0.h[0] = *(const u16x8*)(bp);
        b0.h[1] = *(const u16x8*)(bp + 8);
        b1.h[0] = *(const u16x8*)(bp + 512);
        b1.h[1] = *(const u16x8*)(bp + 520);
        v8f acc2 = {0.f, 0.f, 0.f, 0.f, 0.f, 0.f, 0.f, 0.f};
        acc2 = __builtin_amdgcn_wmma_f32_16x16x32_bf16(false, a0.v, false, b0.v,
                                                       (short)0, acc2, false, false);
        acc2 = __builtin_amdgcn_wmma_f32_16x16x32_bf16(false, a1.v, false, b1.v,
                                                       (short)0, acc2, false, false);
        if (ln < COUT) {
            float sc2 = s2[ln], sf2 = sh2[ln];
            #pragma unroll
            for (int j = 0; j < 8; ++j) {
                int m  = j + (khalf << 3);
                int xx = x0 + m;
                yout[((long)(b * COUT + ln) * H2 + y) * W2 + xx] =
                    fmaf(acc2[j], sc2, sf2);
            }
        }
    }
}

// ---------------- reserve filter + zero pad (2) ----------------
__global__ __launch_bounds__(256) void thresh_pad_kernel(
    const float* __restrict__ yout, float* __restrict__ xfp)
{
    long i = (long)blockIdx.x * 256 + threadIdx.x;
    const long n = (long)Bv * COUT * XPH * XPW;
    if (i >= n) return;
    int xx = (int)(i % XPW);
    long r = i / XPW;
    int yy = (int)(r % XPH);
    long plane = r / XPH;
    float v = 0.f;
    if (xx >= 2 && xx < XPW - 2 && yy >= 2 && yy < XPH - 2) {
        float t = yout[(plane << 16) + (long)(yy - 2) * W2 + (xx - 2)];
        v = (t > 1.0f) ? t : 0.f;
    }
    xfp[i] = v;
}

// ---------------- 5x5 sliding window sum ----------------
__global__ __launch_bounds__(256) void winsum_kernel(
    const float* __restrict__ xfp, float* __restrict__ wsb)
{
    long i = (long)blockIdx.x * 256 + threadIdx.x;
    const long n = (long)Bv * COUT * H2 * W2;
    if (i >= n) return;
    int xx = (int)(i & 255);
    int yy = (int)((i >> 8) & 255);
    long plane = i >> 16;
    const float* p = xfp + plane * XPH * XPW;
    float s = 0.f;
    #pragma unroll
    for (int dy = 0; dy < KWIN; ++dy)
        #pragma unroll
        for (int dx = 0; dx < KWIN; ++dx)
            s += p[(yy + dy) * XPW + xx + dx];
    wsb[i] = s;
}

// ---------------- NMS: keep ws only where it equals the 5x5 max ----------------
__global__ __launch_bounds__(256) void nms_kernel(
    const float* __restrict__ wsb, float* __restrict__ msk)
{
    long i = (long)blockIdx.x * 256 + threadIdx.x;
    const long n = (long)Bv * COUT * H2 * W2;
    if (i >= n) return;
    int xx = (int)(i & 255);
    int yy = (int)((i >> 8) & 255);
    long plane = i >> 16;
    const float* p = wsb + (plane << 16);
    float wv = wsb[i];
    float mp = -__builtin_inff();
    for (int dy = -2; dy <= 2; ++dy) {
        int y2 = yy + dy;
        if ((unsigned)y2 >= (unsigned)H2) continue;
        for (int dx = -2; dx <= 2; ++dx) {
            int x2 = xx + dx;
            if ((unsigned)x2 >= (unsigned)W2) continue;
            mp = fmaxf(mp, p[y2 * W2 + x2]);
        }
    }
    msk[i] = (wv == mp) ? wv : 0.f;
}

// ---------------- per-plane top-8 (stable: ties -> lowest index) + positions --------
__global__ __launch_bounds__(256) void topk_kernel(
    const float* __restrict__ msk, int* __restrict__ idxb, float* __restrict__ pos_out)
{
    __shared__ float sv[256 * NUMK];
    __shared__ int   si[256 * NUMK];
    int plane = blockIdx.x;
    int tid = threadIdx.x;
    const float* p = msk + (long)plane * H2 * W2;

    float tv[NUMK]; int ti[NUMK];
    #pragma unroll
    for (int k = 0; k < NUMK; ++k) { tv[k] = -__builtin_inff(); ti[k] = 0x7fffffff; }
    for (int i = tid; i < H2 * W2; i += 256) {
        float v = p[i];
        if (v > tv[NUMK - 1] || (v == tv[NUMK - 1] && i < ti[NUMK - 1])) {
            int q = NUMK - 1;
            while (q > 0 && (v > tv[q - 1] || (v == tv[q - 1] && i < ti[q - 1]))) {
                tv[q] = tv[q - 1]; ti[q] = ti[q - 1]; --q;
            }
            tv[q] = v; ti[q] = i;
        }
    }
    #pragma unroll
    for (int k = 0; k < NUMK; ++k) { sv[tid * NUMK + k] = tv[k]; si[tid * NUMK + k] = ti[k]; }

    for (int stride = 128; stride > 0; stride >>= 1) {
        __syncthreads();
        if (tid < stride) {
            float mv[NUMK]; int mi[NUMK];
            int ia = 0, ib = 0;
            float* av = &sv[tid * NUMK];            int* ai = &si[tid * NUMK];
            float* bv = &sv[(tid + stride) * NUMK]; int* bi = &si[(tid + stride) * NUMK];
            #pragma unroll
            for (int o = 0; o < NUMK; ++o) {
                bool ta = (av[ia] > bv[ib]) || (av[ia] == bv[ib] && ai[ia] < bi[ib]);
                if (ta) { mv[o] = av[ia]; mi[o] = ai[ia]; ++ia; }
                else    { mv[o] = bv[ib]; mi[o] = bi[ib]; ++ib; }
            }
            #pragma unroll
            for (int o = 0; o < NUMK; ++o) { av[o] = mv[o]; ai[o] = mi[o]; }
        }
    }
    __syncthreads();
    if (tid < NUMK) {
        int id = si[tid];
        idxb[plane * NUMK + tid] = id;
        int h = id >> 8, w = id & 255;
        int x1 = w - 2; x1 = x1 < 0 ? 0 : (x1 > W2 - 1 ? W2 - 1 : x1);
        int y1 = h - 2; y1 = y1 < 0 ? 0 : (y1 > H2 - 1 ? H2 - 1 : y1);
        int x2 = w + 2; x2 = x2 < 0 ? 0 : (x2 > W2 - 1 ? W2 - 1 : x2);
        int y2 = h + 2; y2 = y2 < 0 ? 0 : (y2 > H2 - 1 ? H2 - 1 : y2);
        float* o = pos_out + ((long)plane * NUMK + tid) * 4;
        o[0] = (float)x1; o[1] = (float)y1; o[2] = (float)x2; o[3] = (float)y2;
    }
}

// ---------------- extract 5x5 patches from padded xf ----------------
__global__ __launch_bounds__(256) void patches_kernel(
    const float* __restrict__ xfp, const int* __restrict__ idxb,
    float* __restrict__ out)
{
    int i = blockIdx.x * 256 + threadIdx.x;
    const int n = Bv * COUT * NUMK * KWIN * KWIN;
    if (i >= n) return;
    int dx = i % KWIN;
    int dy = (i / KWIN) % KWIN;
    int nn = (i / (KWIN * KWIN)) % NUMK;
    int plane = i / (NUMK * KWIN * KWIN);
    int id = idxb[plane * NUMK + nn];
    int h = id >> 8, w = id & 255;
    out[i] = xfp[(long)plane * XPH * XPW + (h + dy) * XPW + (w + dx)];
}

// ---------------- launch ----------------
extern "C" void kernel_launch(void* const* d_in, const int* in_sizes, int n_in,
                              void* d_out, int out_size, void* d_ws, size_t ws_size,
                              hipStream_t stream)
{
    const float* x   = (const float*)d_in[0];
    const float* w1  = (const float*)d_in[1];
    const float* b1  = (const float*)d_in[2];
    const float* g1  = (const float*)d_in[3];
    const float* be1 = (const float*)d_in[4];
    const float* m1  = (const float*)d_in[5];
    const float* v1  = (const float*)d_in[6];
    const float* w2  = (const float*)d_in[7];
    const float* g2  = (const float*)d_in[8];
    const float* be2 = (const float*)d_in[9];
    const float* m2  = (const float*)d_in[10];
    const float* v2  = (const float*)d_in[11];
    float* out = (float*)d_out;

    char* ws = (char*)d_ws;
    size_t off = 0;
    auto take = [&](size_t bytes) -> char* {
        char* pp = ws + off;
        off = (off + bytes + 255) & ~(size_t)255;
        return pp;
    };
    unsigned short* up  = (unsigned short*)take((size_t)Bv * UPH * UPW * CIN * 2);
    float* yout = (float*)take((size_t)Bv * COUT * H2 * W2 * 4);
    float* xfp  = (float*)take((size_t)Bv * COUT * XPH * XPW * 4);
    float* wsb  = (float*)take((size_t)Bv * COUT * H2 * W2 * 4);
    float* msk  = (float*)take((size_t)Bv * COUT * H2 * W2 * 4);
    unsigned short* wp1 = (unsigned short*)take(9 * 4 * 2 * 32 * 16 * 2);
    unsigned short* wp2 = (unsigned short*)take(2 * 32 * 16 * 2);
    float* s1  = (float*)take(CIN * 4);
    float* sh1 = (float*)take(CIN * 4);
    float* s2  = (float*)take(COUT * 4);
    float* sh2 = (float*)take(COUT * 4);
    int* idxb  = (int*)take(Bv * COUT * NUMK * 4);

    const int nprep = 9 * 4 * 2 * 32 * 16 + 2 * 32 * 16 + CIN + COUT;
    prep_kernel<<<(nprep + 255) / 256, 256, 0, stream>>>(
        w1, b1, g1, be1, m1, v1, w2, g2, be2, m2, v2, wp1, wp2, s1, sh1, s2, sh2);

    const long nup = (long)Bv * UPH * UPW * CIN;
    upsample_kernel<<<(unsigned)((nup + 255) / 256), 256, 0, stream>>>(x, up);

    conv_fused_kernel<<<Bv * 256 * 8, 256, 0, stream>>>(
        up, wp1, wp2, s1, sh1, s2, sh2, yout);

    const long nxf = (long)Bv * COUT * XPH * XPW;
    thresh_pad_kernel<<<(unsigned)((nxf + 255) / 256), 256, 0, stream>>>(yout, xfp);

    const long nys = (long)Bv * COUT * H2 * W2;
    winsum_kernel<<<(unsigned)((nys + 255) / 256), 256, 0, stream>>>(xfp, wsb);
    nms_kernel<<<(unsigned)((nys + 255) / 256), 256, 0, stream>>>(wsb, msk);

    topk_kernel<<<Bv * COUT, 256, 0, stream>>>(msk, idxb,
        out + (size_t)Bv * COUT * NUMK * KWIN * KWIN);

    const int npat = Bv * COUT * NUMK * KWIN * KWIN;
    patches_kernel<<<(npat + 255) / 256, 256, 0, stream>>>(xfp, idxb, out);
}
